// GatedSoftmaxAttention_16862041604484
// MI455X (gfx1250) — compile-verified, matrix operand order checked by reading, not run on hardware
//
#include <hip/hip_runtime.h>
#include <hip/hip_bf16.h>

typedef __attribute__((ext_vector_type(16))) __bf16 v16bf;
typedef __attribute__((ext_vector_type(8)))  __bf16 v8bf;
typedef __attribute__((ext_vector_type(8)))  float  v8f;

#define B_   2
#define N_   2048
#define HID_ 896
#define HQ_  14
#define HKV_ 2
#define D_   64
#define GQA_ (HQ_ / HKV_)       // 7
#define M_   (B_ * N_)          // 4096
#define KV_  (HKV_ * D_)        // 128
#define QD_  (HQ_ * D_)         // 896
#define THETA_ 1000000.0f

// ---------------------------------------------------------------------------
// CDNA5 16-bit A/B fragment (16x32): a lane's 16 bf16 elements are two
// contiguous K-runs: k in [hi*8, hi*8+8) and [16+hi*8, 16+hi*8+8).
// With K-runs contiguous + 16B-aligned in LDS, one fragment = 2x ds_load_b128.
// C/D (f32): row m = r + 8*hi, col n = lane&15.
// ---------------------------------------------------------------------------
union FragU { v16bf f; v8bf h[2]; };

__device__ __forceinline__ v16bf frag_from_lds(const __bf16* rowBase, int hi) {
    FragU u;
    u.h[0] = *reinterpret_cast<const v8bf*>(rowBase + hi * 8);
    u.h[1] = *reinterpret_cast<const v8bf*>(rowBase + 16 + hi * 8);
    return u.f;
}

__device__ __forceinline__ v8bf cvt8(float4 a, float4 b) {
    v8bf r;
    r[0] = (__bf16)a.x; r[1] = (__bf16)a.y; r[2] = (__bf16)a.z; r[3] = (__bf16)a.w;
    r[4] = (__bf16)b.x; r[5] = (__bf16)b.y; r[6] = (__bf16)b.z; r[7] = (__bf16)b.w;
    return r;
}
__device__ __forceinline__ float4 ld4(const float* p) {
    return *reinterpret_cast<const float4*>(p);
}

// ---------------------------------------------------------------------------
// Tiled GEMM: C[M,Nc] = A[M,K] @ W[K,Nc] (+bias). fp32 in/out, bf16 WMMA,
// f32 accumulation. 128 threads = 4 waves; wave owns a 16x64 slab.
// Assumes M%64==0, K%32==0, Nc%64==0 (true for all five calls).
// ---------------------------------------------------------------------------
__global__ __launch_bounds__(128)
void gemm_bias_wmma(const float* __restrict__ A, const float* __restrict__ W,
                    const float* __restrict__ bias, float* __restrict__ C,
                    int M, int K, int Nc)
{
    __shared__ alignas(16) __bf16 sA [64][40];  // [row][k]  pitch 80B
    __shared__ alignas(16) __bf16 sWt[64][40];  // [col][k]  (transposed W tile)

    const int lane = threadIdx.x & 31;
    const int wave = threadIdx.x >> 5;
    const int hi   = lane >> 4;
    const int nl   = lane & 15;
    const int rowBase = blockIdx.x * 64;
    const int colBase = blockIdx.y * 64;

    v8f acc[4] = {v8f{}, v8f{}, v8f{}, v8f{}};

    for (int k0 = 0; k0 < K; k0 += 32) {
        // A tile 64x32: task = (row, 8-wide k-group); global float4 coalesced,
        // LDS writes are b128.
        for (int i = threadIdx.x; i < 256; i += 128) {
            int r = i >> 2, g = i & 3;
            const float* src = A + (size_t)(rowBase + r) * K + k0 + g * 8;
            *reinterpret_cast<v8bf*>(&sA[r][g * 8]) = cvt8(ld4(src), ld4(src + 4));
        }
        // W tile 32x64 stored transposed: task = (col n, 8-deep k-group);
        // consecutive threads -> consecutive n, so each k-row read coalesces.
        for (int i = threadIdx.x; i < 256; i += 128) {
            int n = i & 63, j = i >> 6;
            const float* src = W + (size_t)(k0 + j * 8) * Nc + colBase + n;
            v8bf w;
            #pragma unroll
            for (int jj = 0; jj < 8; ++jj)
                w[jj] = (__bf16)src[(size_t)jj * Nc];
            *reinterpret_cast<v8bf*>(&sWt[n][j * 8]) = w;
        }
        __syncthreads();

        v16bf afrag = frag_from_lds(&sA[wave * 16 + nl][0], hi);
        #pragma unroll
        for (int t = 0; t < 4; ++t) {
            v16bf bfrag = frag_from_lds(&sWt[t * 16 + nl][0], hi);
            acc[t] = __builtin_amdgcn_wmma_f32_16x16x32_bf16(
                false, afrag, false, bfrag, (short)0, acc[t], false, false);
        }
        __syncthreads();
    }

    #pragma unroll
    for (int t = 0; t < 4; ++t) {
        int gn = colBase + t * 16 + nl;
        float bb = bias ? bias[gn] : 0.0f;
        #pragma unroll
        for (int r = 0; r < 8; ++r) {
            int gm = rowBase + wave * 16 + r + 8 * hi;
            C[(size_t)gm * Nc + gn] = acc[t][r] + bb;
        }
    }
}

// ---------------------------------------------------------------------------
// RoPE on Q [M_,896] and K [M_,128] in place (rotate-half convention).
// ---------------------------------------------------------------------------
__global__ __launch_bounds__(128)
void rope_kernel(float* __restrict__ Q, float* __restrict__ Kc,
                 const long long* __restrict__ pos)
{
    const int tok = blockIdx.x;
    const float p = (float)pos[tok];

    for (int i = threadIdx.x; i < HQ_ * (D_ / 2); i += blockDim.x) {
        int h = i / (D_ / 2), d = i % (D_ / 2);
        float ang = p * __powf(THETA_, -((float)(2 * d) / (float)D_));
        float c = __cosf(ang), s = __sinf(ang);
        float* q = Q + (size_t)tok * QD_ + h * D_;
        float x0 = q[d], x1 = q[d + D_ / 2];
        q[d]          = x0 * c - x1 * s;
        q[d + D_ / 2] = x1 * c + x0 * s;
    }
    for (int i = threadIdx.x; i < HKV_ * (D_ / 2); i += blockDim.x) {
        int h = i / (D_ / 2), d = i % (D_ / 2);
        float ang = p * __powf(THETA_, -((float)(2 * d) / (float)D_));
        float c = __cosf(ang), s = __sinf(ang);
        float* k = Kc + (size_t)tok * KV_ + h * D_;
        float x0 = k[d], x1 = k[d + D_ / 2];
        k[d]          = x0 * c - x1 * s;
        k[d + D_ / 2] = x1 * c + x0 * s;
    }
}

// ---------------------------------------------------------------------------
// Transpose V: VT[(b*128 + c) * N_ + n] = V[(b*N_ + n) * 128 + c]
// so PV B-fragments become contiguous K(=key)-runs.
// ---------------------------------------------------------------------------
__global__ __launch_bounds__(256)
void transpose_v(const float* __restrict__ Vin, float* __restrict__ VT)
{
    __shared__ float tile[32][33];
    const int b  = blockIdx.z;
    const int n0 = blockIdx.x * 32;
    const int c0 = blockIdx.y * 32;
    const int tx = threadIdx.x, ty = threadIdx.y;   // 32 x 8
    for (int yy = ty; yy < 32; yy += 8)
        tile[yy][tx] = Vin[(size_t)(b * N_ + n0 + yy) * KV_ + c0 + tx];
    __syncthreads();
    for (int yy = ty; yy < 32; yy += 8)
        VT[(size_t)(b * KV_ + c0 + yy) * N_ + n0 + tx] = tile[tx][yy];
}

// ---------------------------------------------------------------------------
// Flash attention: one wave32 per (16-query tile, b, h). Causal, 32-key
// blocks through LDS. 4 WMMAs for S, online softmax, 4 WMMAs for PV.
// Sigmoid gate fused in the epilogue.
// ---------------------------------------------------------------------------
__global__ __launch_bounds__(32)
void attn_kernel(const float* __restrict__ Q, const float* __restrict__ Kc,
                 const float* __restrict__ VT, const float* __restrict__ G,
                 float* __restrict__ O)
{
    __shared__ alignas(16) __bf16 sK [32][72];  // [key][d]   pitch 144B
    __shared__ alignas(16) __bf16 sVt[64][40];  // [d][key]   pitch 80B
    __shared__ alignas(16) __bf16 sP [16][40];  // [qrow][key]

    const int qtile = blockIdx.x;
    const int bh    = blockIdx.y;
    const int b     = bh / HQ_;
    const int h     = bh % HQ_;
    const int kv    = h / GQA_;
    const int lane  = threadIdx.x;
    const int hi    = lane >> 4;
    const int nl    = lane & 15;
    const int qBase = qtile * 16;
    const float scale = 0.125f;   // 1/sqrt(64)

    // Q fragments: A-layout, two K=32 chunks of D=64, straight from global
    // (two contiguous 8-float runs per chunk half).
    v16bf aq[2];
    {
        const float* qrow = Q + (size_t)(b * N_ + qBase + nl) * QD_ + h * D_;
        #pragma unroll
        for (int c = 0; c < 2; ++c) {
            const float* base = qrow + c * 32;
            FragU u;
            u.h[0] = cvt8(ld4(base + hi * 8),      ld4(base + hi * 8 + 4));
            u.h[1] = cvt8(ld4(base + 16 + hi * 8), ld4(base + 16 + hi * 8 + 4));
            aq[c] = u.f;
        }
    }

    v8f   acc[4] = {v8f{}, v8f{}, v8f{}, v8f{}};
    float rowm[8], rowl[8];
    #pragma unroll
    for (int r = 0; r < 8; ++r) { rowm[r] = -INFINITY; rowl[r] = 0.0f; }

    const int kbEnd = qBase + 16;   // exclusive causal key limit
    for (int kb = 0; kb < kbEnd; kb += 32) {
        // K block: lane-per-key row, b128 reads + b128 LDS writes.
        {
            const float* krow = Kc + (size_t)(b * N_ + kb + lane) * KV_ + kv * D_;
            #pragma unroll
            for (int g = 0; g < 8; ++g)
                *reinterpret_cast<v8bf*>(&sK[lane][g * 8]) =
                    cvt8(ld4(krow + g * 8), ld4(krow + g * 8 + 4));
        }
        // V block from VT: lane handles d = lane and lane+32; 32 contiguous
        // keys per row, b128 reads + b128 LDS writes.
        #pragma unroll
        for (int half = 0; half < 2; ++half) {
            int d = lane + half * 32;
            const float* vrow = VT + (size_t)(b * KV_ + kv * D_ + d) * N_ + kb;
            #pragma unroll
            for (int g = 0; g < 4; ++g)
                *reinterpret_cast<v8bf*>(&sVt[d][g * 8]) =
                    cvt8(ld4(vrow + g * 8), ld4(vrow + g * 8 + 4));
        }
        if (kb + 32 < kbEnd) {   // prefetch next block (global_prefetch_b8)
            __builtin_prefetch(
                Kc + (size_t)(b * N_ + kb + 32 + lane) * KV_ + kv * D_, 0, 1);
            __builtin_prefetch(
                VT + (size_t)(b * KV_ + kv * D_ + lane) * N_ + kb + 32, 0, 1);
        }
        __syncthreads();

        // S = Q @ K^T : two 16x16 n-tiles x two K=32 chunks.
        v8f sfrag[2] = {v8f{}, v8f{}};
        #pragma unroll
        for (int t = 0; t < 2; ++t) {
            #pragma unroll
            for (int c = 0; c < 2; ++c) {
                v16bf bfrag = frag_from_lds(&sK[t * 16 + nl][c * 32], hi);
                sfrag[t] = __builtin_amdgcn_wmma_f32_16x16x32_bf16(
                    false, aq[c], false, bfrag, (short)0, sfrag[t], false, false);
            }
        }

        // online softmax (row in VGPR index, col in lane)
        float sv[2][8];
        #pragma unroll
        for (int t = 0; t < 2; ++t)
            #pragma unroll
            for (int r = 0; r < 8; ++r) {
                int m   = r + 8 * hi;
                int key = kb + t * 16 + nl;
                sv[t][r] = (key <= qBase + m) ? sfrag[t][r] * scale : -INFINITY;
            }

        #pragma unroll
        for (int r = 0; r < 8; ++r) {
            float bm = fmaxf(sv[0][r], sv[1][r]);
            #pragma unroll
            for (int off = 8; off > 0; off >>= 1)
                bm = fmaxf(bm, __shfl_xor(bm, off, 32));
            float mnew  = fmaxf(rowm[r], bm);
            float alpha = __expf(rowm[r] - mnew);
            rowm[r] = mnew;

            float p0 = (sv[0][r] == -INFINITY) ? 0.0f : __expf(sv[0][r] - mnew);
            float p1 = (sv[1][r] == -INFINITY) ? 0.0f : __expf(sv[1][r] - mnew);
            float psum = p0 + p1;
            #pragma unroll
            for (int off = 8; off > 0; off >>= 1)
                psum += __shfl_xor(psum, off, 32);
            rowl[r] = rowl[r] * alpha + psum;

            #pragma unroll
            for (int t = 0; t < 4; ++t)
                acc[t][r] = acc[t][r] * alpha;

            int m = r + 8 * hi;
            sP[m][nl]      = (__bf16)p0;
            sP[m][16 + nl] = (__bf16)p1;
        }
        __syncthreads();

        // O += P @ V : A frag from sP, B frags = contiguous key-runs of sVt.
        v16bf ap = frag_from_lds(&sP[nl][0], hi);
        #pragma unroll
        for (int t = 0; t < 4; ++t) {
            v16bf bfv = frag_from_lds(&sVt[t * 16 + nl][0], hi);
            acc[t] = __builtin_amdgcn_wmma_f32_16x16x32_bf16(
                false, ap, false, bfv, (short)0, acc[t], false, false);
        }
        __syncthreads();   // before next block overwrites sK/sVt
    }

    // epilogue: normalize, sigmoid gate, store gated O
    #pragma unroll
    for (int t = 0; t < 4; ++t) {
        #pragma unroll
        for (int r = 0; r < 8; ++r) {
            int m = r + 8 * hi;
            size_t idx = (size_t)(b * N_ + qBase + m) * QD_ + h * D_ + t * 16 + nl;
            float o    = acc[t][r] / rowl[r];
            float gate = 1.0f / (1.0f + __expf(-G[idx]));
            O[idx] = o * gate;
        }
    }
}

// ---------------------------------------------------------------------------
// Launch
// ---------------------------------------------------------------------------
extern "C" void kernel_launch(void* const* d_in, const int* in_sizes, int n_in,
                              void* d_out, int out_size, void* d_ws, size_t ws_size,
                              hipStream_t stream)
{
    (void)in_sizes; (void)n_in; (void)out_size; (void)ws_size;

    const float*     hidden = (const float*)d_in[0];
    const long long* pos    = (const long long*)d_in[1];
    const float*     Wq     = (const float*)d_in[2];
    const float*     bq     = (const float*)d_in[3];
    const float*     Wk     = (const float*)d_in[4];
    const float*     bk     = (const float*)d_in[5];
    const float*     Wv     = (const float*)d_in[6];
    const float*     bv     = (const float*)d_in[7];
    const float*     Wg     = (const float*)d_in[8];
    const float*     Wo     = (const float*)d_in[9];
    float*           out    = (float*)d_out;

    float* Qbuf = (float*)d_ws;                   // 4096 x 896
    float* Kbuf = Qbuf + (size_t)M_ * QD_;        // 4096 x 128
    float* Vbuf = Kbuf + (size_t)M_ * KV_;        // 4096 x 128
    float* Gbuf = Vbuf + (size_t)M_ * KV_;        // 4096 x 896
    float* Obuf = Gbuf + (size_t)M_ * QD_;        // 4096 x 896
    float* VTb  = Obuf + (size_t)M_ * QD_;        // 2 x 128 x 2048

    dim3 blkGemm(128);
    dim3 gQ(M_ / 64, QD_ / 64);    // 64 x 14
    dim3 gK(M_ / 64, KV_ / 64);    // 64 x 2

    gemm_bias_wmma<<<gQ, blkGemm, 0, stream>>>(hidden, Wq, bq, Qbuf, M_, HID_, QD_);
    gemm_bias_wmma<<<gK, blkGemm, 0, stream>>>(hidden, Wk, bk, Kbuf, M_, HID_, KV_);
    gemm_bias_wmma<<<gK, blkGemm, 0, stream>>>(hidden, Wv, bv, Vbuf, M_, HID_, KV_);
    gemm_bias_wmma<<<gQ, blkGemm, 0, stream>>>(hidden, Wg, nullptr, Gbuf, M_, HID_, QD_);

    rope_kernel<<<dim3(M_), dim3(128), 0, stream>>>(Qbuf, Kbuf, pos);

    transpose_v<<<dim3(N_ / 32, KV_ / 32, B_), dim3(32, 8), 0, stream>>>(Vbuf, VTb);

    attn_kernel<<<dim3(N_ / 16, B_ * HQ_), dim3(32), 0, stream>>>(
        Qbuf, Kbuf, VTb, Gbuf, Obuf);

    gemm_bias_wmma<<<gQ, blkGemm, 0, stream>>>(Obuf, Wo, nullptr, out, M_, QD_, HID_);
}
